// GImodel_54589034332346
// MI455X (gfx1250) — matrix operand order repaired and checked by reading, not compiled, not run on hardware
//
#include <hip/hip_runtime.h>

#define BSz   128
#define SEQL  256
#define NND   22
#define FFD   6
#define HD    128
#define SD    32
#define DDIM  132
#define RROWS 2816      // BSz*NND
#define OUTBLKS 66      // (BSz*DDIM)/256
#define EPSC  1.1920929e-07f

typedef __bf16 bf16_t;
typedef __attribute__((ext_vector_type(16))) __bf16 v16bf;
typedef __attribute__((ext_vector_type(8)))  float  v8f;

__device__ __forceinline__ bf16_t f2bf(float x) { return (bf16_t)x; }  // native cvt (RNE)
__device__ __forceinline__ float sigm(float x) { return 1.f / (1.f + expf(-x)); }
__device__ __forceinline__ float softplus_(float x) { return (x > 20.f) ? x : log1pf(expf(x)); }
__device__ __forceinline__ v8f vz8() {
  v8f z;
#pragma unroll
  for (int i = 0; i < 8; ++i) z[i] = 0.f;
  return z;
}

// CDNA5 async DMA: copy 16B global -> LDS (ASYNCcnt-tracked). lds_off is a byte
// offset from the workgroup LDS base (kernels using this have a single __shared__ block).
__device__ __forceinline__ void async_g2l16(unsigned lds_off, const void* g) {
  asm volatile("global_load_async_to_lds_b128 %0, %1, off" :: "v"(lds_off), "v"(g) : "memory");
}
__device__ __forceinline__ void async_wait0() { asm volatile("s_wait_asynccnt 0" ::: "memory"); }
__device__ __forceinline__ void async_wait1() { asm volatile("s_wait_asynccnt 1" ::: "memory"); }

// acc += A(16 x 32*KT, bf16 in LDS, row stride lda) * Bpacked[nt] (32*KT x 16)
__device__ __forceinline__ v8f wmma_acc(const bf16_t* sA, int lda,
                                        const bf16_t* __restrict__ Bp,
                                        int nt, int KT, int lane, v8f acc) {
  int m  = lane & 15;
  int hi = lane >> 4;
  for (int kt = 0; kt < KT; ++kt) {
    v16bf a, b;
    int ko = kt * 32 + (hi ? 16 : 0);
    const bf16_t* ar = sA + m * lda + ko;
#pragma unroll
    for (int i = 0; i < 8; ++i) { a[i] = ar[i]; a[8 + i] = ar[8 + i]; }
    const bf16_t* br = Bp + ((size_t)(nt * KT + kt) * 32 + lane) * 16;
#pragma unroll
    for (int i = 0; i < 16; ++i) b[i] = br[i];
    acc = __builtin_amdgcn_wmma_f32_16x16x32_bf16(false, a, false, b, (short)0, acc, false, false);
  }
  return acc;
}

// ---------------- prep kernels ----------------
__global__ __launch_bounds__(256) void zero_kernel(float* p, size_t n) {
  size_t i = (size_t)blockIdx.x * 256 + threadIdx.x;
  if (i < n) p[i] = 0.f;
}

// Pack W rows [n][src_off + k] (k in [0,ind)) into WMMA-B fragment order at padded-K
// offset k0, fragments kt in [ktA, ktA+ktN) of KT total. fold: add W[n][fold_off + src_off + k].
__global__ __launch_bounds__(256) void pack_kernel(const float* __restrict__ W, bf16_t* __restrict__ P,
    int outd, int ind, int rowstride, int k0, int ktA, int ktN, int KT, int fold,
    int src_off, int fold_off) {
  int idx = blockIdx.x * 256 + threadIdx.x;
  int total = (outd >> 4) * ktN * 32;
  if (idx >= total) return;
  int lane = idx & 31;
  int q = idx >> 5;
  int kt = ktA + (q % ktN);
  int nt = q / ktN;
  int n = nt * 16 + (lane & 15);
  bf16_t* dst = P + ((size_t)(nt * KT + kt) * 32 + lane) * 16;
  int kbase = kt * 32 + ((lane >> 4) * 16);
#pragma unroll
  for (int i = 0; i < 16; ++i) {
    int k = kbase + i - k0;
    float v = 0.f;
    if (k >= 0 && k < ind) {
      v = W[(size_t)n * rowstride + src_off + k];
      if (fold) v += W[(size_t)n * rowstride + fold_off + src_off + k];
    }
    dst[i] = f2bf(v);
  }
}

// ---------------- per-step kernels ----------------
// Fused 2-layer LSTM for a 16-row tile. States h1/h2 live as bf16 (GEMM-only consumers);
// c1/c2 stay fp32. h-tiles staged into LDS via async DMA; old-h2 prefetch overlaps GEMM0.
// W0p: [Wih0 (K 0..5) ; Whh0 (K 32..159)], KT=5.  W1p: [Wih1 ; Whh1], KT=8.
__global__ __launch_bounds__(256) void lstm_kernel(const float* __restrict__ prev_out,
    float* __restrict__ c1, float* __restrict__ c2,
    bf16_t* __restrict__ h1b, bf16_t* __restrict__ h2b,
    const bf16_t* __restrict__ W0p, const bf16_t* __restrict__ W1p,
    const float* __restrict__ bih0, const float* __restrict__ bhh0,
    const float* __restrict__ bih1, const float* __restrict__ bhh1) {
  __shared__ __align__(16) char smem[5120 + 8192 + 32768];
  bf16_t* sA0 = (bf16_t*)smem;              // 16 x 160 (LDS bytes [0,5120))
  bf16_t* sA1 = (bf16_t*)(smem + 5120);     // 16 x 256 ([5120,13312))
  float*  sG  = (float*)(smem + 13312);     // 16 x 512 fp32
  int tid = threadIdx.x, lane = tid & 31, wave = tid >> 5;
  int row0 = blockIdx.x * 16;
  {
    int r = tid >> 4, ch = tid & 15;
    // h1 -> A0 cols 32..159 (row stride 320B)
    async_g2l16((unsigned)(r * 320 + 64 + ch * 16),
                (const char*)(h1b + (size_t)(row0 + r) * HD) + ch * 16);
    // old h2 -> A1 cols 128..255 (row stride 512B): prefetch, consumed after GEMM0
    async_g2l16((unsigned)(5120 + r * 512 + 256 + ch * 16),
                (const char*)(h2b + (size_t)(row0 + r) * HD) + ch * 16);
  }
  // x into A0 cols 0..5, zero-pad 6..31
  for (int i = tid; i < 16 * 32; i += 256) {
    int r = i >> 5, k = i & 31;
    float v = (k < FFD) ? prev_out[(size_t)(row0 + r) * FFD + k] : 0.f;
    sA0[r * 160 + k] = f2bf(v);
  }
  async_wait1();  // h1 copy landed (async loads complete in order)
  __syncthreads();
  for (int jj = 0; jj < 4; ++jj) {
    int nt = wave * 4 + jj;
    v8f acc = vz8();
    acc = wmma_acc(sA0, 160, W0p, nt, 5, lane, acc);
    int col = nt * 16 + (lane & 15);
    float bias = bih0[col] + bhh0[col];
    int mb = (lane >= 16) ? 8 : 0;
#pragma unroll
    for (int j = 0; j < 8; ++j) sG[(mb + j) * 512 + col] = acc[j] + bias;
  }
  __syncthreads();
  // layer-0 pointwise; h1new -> global bf16 + A1 cols 0..127
  for (int i = tid; i < 16 * 128; i += 256) {
    int r = i >> 7, hh = i & 127;
    float gi = sG[r * 512 + hh], gf = sG[r * 512 + 128 + hh];
    float gg = sG[r * 512 + 256 + hh], go = sG[r * 512 + 384 + hh];
    size_t gidx = (size_t)(row0 + r) * HD + hh;
    float cn = sigm(gf) * c1[gidx] + sigm(gi) * tanhf(gg);
    float hn = sigm(go) * tanhf(cn);
    c1[gidx] = cn;
    bf16_t hb = f2bf(hn);
    h1b[gidx] = hb;
    sA1[r * 256 + hh] = hb;
  }
  async_wait0();  // old-h2 prefetch landed
  __syncthreads();
  for (int jj = 0; jj < 4; ++jj) {
    int nt = wave * 4 + jj;
    v8f acc = vz8();
    acc = wmma_acc(sA1, 256, W1p, nt, 8, lane, acc);
    int col = nt * 16 + (lane & 15);
    float bias = bih1[col] + bhh1[col];
    int mb = (lane >= 16) ? 8 : 0;
#pragma unroll
    for (int j = 0; j < 8; ++j) sG[(mb + j) * 512 + col] = acc[j] + bias;
  }
  __syncthreads();
  for (int i = tid; i < 16 * 128; i += 256) {
    int r = i >> 7, hh = i & 127;
    float gi = sG[r * 512 + hh], gf = sG[r * 512 + 128 + hh];
    float gg = sG[r * 512 + 256 + hh], go = sG[r * 512 + 384 + hh];
    size_t gidx = (size_t)(row0 + r) * HD + hh;
    float cn = sigm(gf) * c2[gidx] + sigm(gi) * tanhf(gg);
    c2[gidx] = cn;
    h2b[gidx] = f2bf(sigm(go) * tanhf(cn));
  }
}

// Graphnet edge stage. withTgt=1 (enc): A = [h2(0..127) | tgt(128..133) | pad], lda=160, KT1=5
// (weights packed with matching column permutation). withTgt=0 (prior): A = h2, lda=128, KT1=4.
// h2 tile arrives via async DMA.
__global__ __launch_bounds__(256) void gn_e_kernel(int t, int withTgt,
    const float* __restrict__ target, const bf16_t* __restrict__ h2b,
    const bf16_t* __restrict__ E1p, const float* __restrict__ eb1,
    const bf16_t* __restrict__ E2p, const float* __restrict__ eb2,
    int KT1, int lda, float* __restrict__ e_out) {
  __shared__ __align__(16) char smem[5120 + 4096];
  bf16_t* sA = (bf16_t*)smem;               // 16 x lda ([0,5120))
  bf16_t* sT = (bf16_t*)(smem + 5120);      // 16 x 128
  int tid = threadIdx.x, lane = tid & 31, wave = tid >> 5;
  int row0 = blockIdx.x * 16;
  {
    int r = tid >> 4, ch = tid & 15;
    unsigned loff = (unsigned)(withTgt ? (r * 320 + ch * 16) : (r * 256 + ch * 16));
    async_g2l16(loff, (const char*)(h2b + (size_t)(row0 + r) * HD) + ch * 16);
  }
  if (withTgt) {
    for (int i = tid; i < 16 * 32; i += 256) {
      int r = i >> 5, k = 128 + (i & 31);
      int gr = row0 + r, b = gr / NND, n = gr % NND;
      float v = (k < 134) ? target[((size_t)b * SEQL + t) * DDIM + n * FFD + (k - 128)] : 0.f;
      sA[r * 160 + k] = f2bf(v);
    }
  }
  async_wait0();
  __syncthreads();
  {
    v8f acc = vz8();
    acc = wmma_acc(sA, lda, E1p, wave, KT1, lane, acc);
    int col = wave * 16 + (lane & 15);
    float bias = eb1[col];
    int mb = (lane >= 16) ? 8 : 0;
#pragma unroll
    for (int j = 0; j < 8; ++j) sT[(mb + j) * HD + col] = f2bf(fmaxf(acc[j] + bias, 0.f));
  }
  __syncthreads();
  {
    v8f acc = vz8();
    acc = wmma_acc(sT, HD, E2p, wave, 4, lane, acc);
    int col = wave * 16 + (lane & 15);
    float bias = eb2[col];
    int mb = (lane >= 16) ? 8 : 0;
#pragma unroll
    for (int j = 0; j < 8; ++j) e_out[(size_t)(row0 + mb + j) * HD + col] = acc[j] + bias;
  }
}

// Graphnet node stage per-batch (nd/m/s constant over the N nodes): 16 batch rows per block.
__global__ __launch_bounds__(256) void gn_p2_kernel(const float* __restrict__ e,
    const bf16_t* __restrict__ N1p, const float* __restrict__ nb1,
    const bf16_t* __restrict__ N2p, const float* __restrict__ nb2,
    const bf16_t* __restrict__ Mp,  const float* __restrict__ mbv,
    const bf16_t* __restrict__ Sp,  const float* __restrict__ sbv,
    float* __restrict__ out_m, float* __restrict__ out_s) {
  __shared__ bf16_t sA[16 * 128];
  __shared__ bf16_t sT[16 * 128];
  int tid = threadIdx.x, lane = tid & 31, wave = tid >> 5;
  int b0 = blockIdx.x * 16;
  for (int i = tid; i < 16 * 128; i += 256) {
    int r = i >> 7, k = i & 127;
    const float* ep = e + (size_t)(b0 + r) * NND * HD + k;
    float s = 0.f;
    for (int n = 0; n < NND; ++n) s += ep[(size_t)n * HD];
    sA[r * HD + k] = f2bf(s);
  }
  __syncthreads();
  {
    v8f acc = vz8();
    acc = wmma_acc(sA, HD, N1p, wave, 4, lane, acc);
    int col = wave * 16 + (lane & 15);
    float bias = nb1[col];
    int mb = (lane >= 16) ? 8 : 0;
#pragma unroll
    for (int j = 0; j < 8; ++j) sT[(mb + j) * HD + col] = f2bf(fmaxf(acc[j] + bias, 0.f));
  }
  __syncthreads();
  {
    v8f acc = vz8();
    acc = wmma_acc(sT, HD, N2p, wave, 4, lane, acc);
    int col = wave * 16 + (lane & 15);
    float bias = nb2[col];
    int mb = (lane >= 16) ? 8 : 0;
#pragma unroll
    for (int j = 0; j < 8; ++j) sA[(mb + j) * HD + col] = f2bf(acc[j] + bias);  // nd
  }
  __syncthreads();
  if (wave < 2) {
    v8f acc = vz8();
    acc = wmma_acc(sA, HD, Mp, wave, 4, lane, acc);
    int col = wave * 16 + (lane & 15);
    float bias = mbv[col];
    int mb = (lane >= 16) ? 8 : 0;
#pragma unroll
    for (int j = 0; j < 8; ++j) out_m[(size_t)(b0 + mb + j) * SD + col] = acc[j] + bias;
  } else if (wave < 4) {
    v8f acc = vz8();
    acc = wmma_acc(sA, HD, Sp, wave - 2, 4, lane, acc);
    int col = (wave - 2) * 16 + (lane & 15);
    float bias = sbv[col];
    int mb = (lane >= 16) ? 8 : 0;
#pragma unroll
    for (int j = 0; j < 8; ++j) out_s[(size_t)(b0 + mb + j) * SD + col] = softplus_(acc[j] + bias);
  }
}

// Decoder edge stage, fused z = relu((ee*enc_s+enc_m)@phi^T + phib); A = [z(0..31) | h2(32..159)].
__global__ __launch_bounds__(256) void gn_e_dec_kernel(int t,
    const float* __restrict__ eps_enc, const float* __restrict__ enc_m, const float* __restrict__ enc_s,
    const bf16_t* __restrict__ h2b, const bf16_t* __restrict__ phiP, const float* __restrict__ phib,
    const bf16_t* __restrict__ E1p, const float* __restrict__ eb1,
    const bf16_t* __restrict__ E2p, const float* __restrict__ eb2,
    float* __restrict__ e_out) {
  __shared__ __align__(16) char smem[5120 + 4096 + 1024];
  bf16_t* sA = (bf16_t*)smem;               // 16 x 160 ([0,5120))
  bf16_t* sT = (bf16_t*)(smem + 5120);      // 16 x 128
  bf16_t* sZ = (bf16_t*)(smem + 9216);      // 16 x 32
  int tid = threadIdx.x, lane = tid & 31, wave = tid >> 5;
  int row0 = blockIdx.x * 16;
  {
    int r = tid >> 4, ch = tid & 15;  // h2 -> A cols 32..159
    async_g2l16((unsigned)(r * 320 + 64 + ch * 16),
                (const char*)(h2b + (size_t)(row0 + r) * HD) + ch * 16);
  }
  for (int i = tid; i < 16 * 32; i += 256) {
    int r = i >> 5, s = i & 31;
    int gr = row0 + r, b = gr / NND, n = gr % NND;
    float ee = eps_enc[(((size_t)t * BSz + b) * NND + n) * SD + s];
    sZ[r * 32 + s] = f2bf(ee * enc_s[(size_t)b * SD + s] + enc_m[(size_t)b * SD + s]);
  }
  async_wait0();
  __syncthreads();
  if (wave < 2) {  // z projection 32->32 via WMMA, KT=1
    v8f acc = vz8();
    acc = wmma_acc(sZ, 32, phiP, wave, 1, lane, acc);
    int col = wave * 16 + (lane & 15);
    float bias = phib[col];
    int mb = (lane >= 16) ? 8 : 0;
#pragma unroll
    for (int j = 0; j < 8; ++j) sA[(mb + j) * 160 + col] = f2bf(fmaxf(acc[j] + bias, 0.f));
  }
  __syncthreads();
  {
    v8f acc = vz8();
    acc = wmma_acc(sA, 160, E1p, wave, 5, lane, acc);
    int col = wave * 16 + (lane & 15);
    float bias = eb1[col];
    int mb = (lane >= 16) ? 8 : 0;
#pragma unroll
    for (int j = 0; j < 8; ++j) sT[(mb + j) * HD + col] = f2bf(fmaxf(acc[j] + bias, 0.f));
  }
  __syncthreads();
  {
    v8f acc = vz8();
    acc = wmma_acc(sT, HD, E2p, wave, 4, lane, acc);
    int col = wave * 16 + (lane & 15);
    float bias = eb2[col];
    int mb = (lane >= 16) ? 8 : 0;
#pragma unroll
    for (int j = 0; j < 8; ++j) e_out[(size_t)(row0 + mb + j) * HD + col] = acc[j] + bias;
  }
}

// dx, rep, pred write, prev_out update, deterministic per-block recons partials.
__global__ __launch_bounds__(256) void out_step_kernel(int t,
    const float* __restrict__ eps_dec, const float* __restrict__ dm, const float* __restrict__ ds,
    const float* __restrict__ outW, const float* __restrict__ outb,
    const float* __restrict__ input, const int* __restrict__ mask,
    float* __restrict__ prev_out, float* __restrict__ pred, float* __restrict__ partials) {
  __shared__ float red[256];
  int tid = threadIdx.x;
  int idx = blockIdx.x * 256 + tid;
  float contrib = 0.f;
  if (idx < BSz * DDIM) {
    int b = idx / DDIM, d = idx % DDIM, n = d / FFD, f = d % FFD;
    const float* edp = eps_dec + (((size_t)t * BSz + b) * NND + n) * SD;
    const float* dmp = dm + (size_t)b * SD;
    const float* dsp = ds + (size_t)b * SD;
    const float* wp = outW + (size_t)f * SD;
    float dx = outb[f];
#pragma unroll 8
    for (int s = 0; s < SD; ++s) dx += (edp[s] * dsp[s] + dmp[s]) * wp[s];
    size_t iidx = ((size_t)b * SEQL + t) * DDIM + d;
    float in_t = input[iidx];
    float prev_in = (t == 0) ? input[(size_t)b * SEQL * DDIM + d] : prev_out[idx];
    float rep = (mask[iidx] == 0) ? (prev_in + dx) : in_t;
    pred[iidx] = rep;
    prev_out[idx] = rep;
    float idelta = (t == 0) ? 0.f : (in_t - input[iidx - DDIM]);
    float df = dx - idelta;
    contrib = df * df;
  }
  red[tid] = contrib;
  __syncthreads();
  for (int s = 128; s > 0; s >>= 1) { if (tid < s) red[tid] += red[tid + s]; __syncthreads(); }
  if (tid == 0) partials[t * OUTBLKS + blockIdx.x] = red[0];
}

__global__ __launch_bounds__(256) void final_kernel(const float* __restrict__ partials,
    const float* __restrict__ enc_m, const float* __restrict__ enc_s,
    const float* __restrict__ pri_m, const float* __restrict__ pri_s, float* __restrict__ loss_out) {
  __shared__ float red[256];
  int tid = threadIdx.x;
  float s1 = 0.f;
  for (int i = tid; i < SEQL * OUTBLKS; i += 256) s1 += partials[i];
  red[tid] = s1;
  __syncthreads();
  for (int s = 128; s > 0; s >>= 1) { if (tid < s) red[tid] += red[tid + s]; __syncthreads(); }
  float recons = red[0] / (float)((size_t)BSz * SEQL * DDIM);
  __syncthreads();
  float s2 = 0.f;
  for (int i = tid; i < BSz * SD; i += 256) {
    float es = enc_s[i], ps = pri_s[i], dmn = enc_m[i] - pri_m[i];
    s2 += 2.f * logf(ps + EPSC) - 2.f * logf(es + EPSC) + (es * es + dmn * dmn) / (ps * ps) - 1.f;
  }
  red[tid] = s2;
  __syncthreads();
  for (int s = 128; s > 0; s >>= 1) { if (tid < s) red[tid] += red[tid + s]; __syncthreads(); }
  if (tid == 0) loss_out[0] = recons + 0.001f * 0.5f * (float)NND * red[0];
}

// ---------------- host ----------------
extern "C" void kernel_launch(void* const* d_in, const int* in_sizes, int n_in,
                              void* d_out, int out_size, void* d_ws, size_t ws_size,
                              hipStream_t stream) {
  (void)in_sizes; (void)n_in; (void)out_size; (void)ws_size;
  const float* input   = (const float*)d_in[0];
  const float* target  = (const float*)d_in[1];
  const int*   mask    = (const int*)d_in[2];
  const float* eps_enc = (const float*)d_in[3];
  const float* eps_dec = (const float*)d_in[4];
  const float* Wih0 = (const float*)d_in[5];  const float* bih0 = (const float*)d_in[6];
  const float* Whh0 = (const float*)d_in[7];  const float* bhh0 = (const float*)d_in[8];
  const float* Wih1 = (const float*)d_in[9];  const float* bih1 = (const float*)d_in[10];
  const float* Whh1 = (const float*)d_in[11]; const float* bhh1 = (const float*)d_in[12];
  const float* e_eW1 = (const float*)d_in[13]; const float* e_eb1 = (const float*)d_in[14];
  const float* e_eW2 = (const float*)d_in[15]; const float* e_eb2 = (const float*)d_in[16];
  const float* e_nW1 = (const float*)d_in[17]; const float* e_nb1 = (const float*)d_in[18];
  const float* e_nW2 = (const float*)d_in[19]; const float* e_nb2 = (const float*)d_in[20];
  const float* e_mW  = (const float*)d_in[21]; const float* e_mb  = (const float*)d_in[22];
  const float* e_sW  = (const float*)d_in[23]; const float* e_sb  = (const float*)d_in[24];
  const float* p_eW1 = (const float*)d_in[25]; const float* p_eb1 = (const float*)d_in[26];
  const float* p_eW2 = (const float*)d_in[27]; const float* p_eb2 = (const float*)d_in[28];
  const float* p_nW1 = (const float*)d_in[29]; const float* p_nb1 = (const float*)d_in[30];
  const float* p_nW2 = (const float*)d_in[31]; const float* p_nb2 = (const float*)d_in[32];
  const float* p_mW  = (const float*)d_in[33]; const float* p_mb  = (const float*)d_in[34];
  const float* p_sW  = (const float*)d_in[35]; const float* p_sb  = (const float*)d_in[36];
  const float* dc_eW1 = (const float*)d_in[37]; const float* dc_eb1 = (const float*)d_in[38];
  const float* dc_eW2 = (const float*)d_in[39]; const float* dc_eb2 = (const float*)d_in[40];
  const float* dc_nW1 = (const float*)d_in[41]; const float* dc_nb1 = (const float*)d_in[42];
  const float* dc_nW2 = (const float*)d_in[43]; const float* dc_nb2 = (const float*)d_in[44];
  const float* dc_mW  = (const float*)d_in[45]; const float* dc_mb  = (const float*)d_in[46];
  const float* dc_sW  = (const float*)d_in[47]; const float* dc_sb  = (const float*)d_in[48];
  const float* phiW = (const float*)d_in[49]; const float* phib = (const float*)d_in[50];
  const float* outW = (const float*)d_in[51]; const float* outb = (const float*)d_in[52];

  float* pred = (float*)d_out;
  float* lossp = pred + (size_t)BSz * SEQL * DDIM;

  char* wp = (char*)d_ws;
  auto abf = [&](size_t n) { bf16_t* r = (bf16_t*)wp; wp += ((n * 2 + 255) / 256) * 256; return r; };
  auto af  = [&](size_t n) { float*  r = (float*)wp;  wp += ((n * 4 + 255) / 256) * 256; return r; };

  bf16_t* W0p = abf(32 * 5 * 512);
  bf16_t* W1p = abf(32 * 8 * 512);
  bf16_t* eE1 = abf(8 * 5 * 512); bf16_t* eE2 = abf(8 * 4 * 512);
  bf16_t* eN1 = abf(8 * 4 * 512); bf16_t* eN2 = abf(8 * 4 * 512);
  bf16_t* eM  = abf(2 * 4 * 512); bf16_t* eS  = abf(2 * 4 * 512);
  bf16_t* pE1 = abf(8 * 4 * 512); bf16_t* pE2 = abf(8 * 4 * 512);
  bf16_t* pN1 = abf(8 * 4 * 512); bf16_t* pN2 = abf(8 * 4 * 512);
  bf16_t* pM  = abf(2 * 4 * 512); bf16_t* pS  = abf(2 * 4 * 512);
  bf16_t* dE1 = abf(8 * 5 * 512); bf16_t* dE2 = abf(8 * 4 * 512);
  bf16_t* dN1 = abf(8 * 4 * 512); bf16_t* dN2 = abf(8 * 4 * 512);
  bf16_t* dM  = abf(2 * 4 * 512); bf16_t* dS  = abf(2 * 4 * 512);
  bf16_t* phiP = abf(2 * 1 * 512);

  // zeroed region: c1, c2, prev (fp32) then h1b, h2b (bf16 shadows) - contiguous, 256B multiples
  float* c1 = af((size_t)RROWS * HD);
  float* c2 = af((size_t)RROWS * HD);
  float* prev = af((size_t)BSz * DDIM);
  bf16_t* h1b = abf((size_t)RROWS * HD);
  bf16_t* h2b = abf((size_t)RROWS * HD);
  float* eEnc = af((size_t)RROWS * HD);
  float* ePri = af((size_t)RROWS * HD);
  float* eDec = af((size_t)RROWS * HD);
  float* encM = af(BSz * SD); float* encS = af(BSz * SD);
  float* priM = af(BSz * SD); float* priS = af(BSz * SD);
  float* dmB = af(BSz * SD);  float* dsB = af(BSz * SD);
  float* partials = af((size_t)SEQL * OUTBLKS);

  auto cdiv = [](int a, int b) { return (a + b - 1) / b; };

  // zero c1,c2,prev,h1b,h2b in one pass (h1b+h2b = RROWS*HD bf16 * 2 = RROWS*HD floats)
  size_t zn = (size_t)3 * RROWS * HD + (size_t)BSz * DDIM;
  zero_kernel<<<(unsigned)((zn + 255) / 256), 256, 0, stream>>>(c1, zn);

  // pack weights into WMMA-B fragment order (bf16): (W,P,outd,ind,rowstride,k0,ktA,ktN,KT,fold,src_off,fold_off)
  pack_kernel<<<cdiv(32 * 1 * 32, 256), 256, 0, stream>>>(Wih0, W0p, 512, 6, 6, 0, 0, 1, 5, 0, 0, 6);
  pack_kernel<<<cdiv(32 * 4 * 32, 256), 256, 0, stream>>>(Whh0, W0p, 512, 128, 128, 32, 1, 4, 5, 0, 0, 128);
  pack_kernel<<<cdiv(32 * 4 * 32, 256), 256, 0, stream>>>(Wih1, W1p, 512, 128, 128, 0, 0, 4, 8, 0, 0, 128);
  pack_kernel<<<cdiv(32 * 4 * 32, 256), 256, 0, stream>>>(Whh1, W1p, 512, 128, 128, 128, 4, 4, 8, 0, 0, 128);
  // enc eW1 folded + column-permuted: A = [node | tgt | pad]
  pack_kernel<<<cdiv(8 * 4 * 32, 256), 256, 0, stream>>>(e_eW1, eE1, 128, 128, 268, 0, 0, 4, 5, 1, 6, 134);
  pack_kernel<<<cdiv(8 * 1 * 32, 256), 256, 0, stream>>>(e_eW1, eE1, 128, 6, 268, 128, 4, 1, 5, 1, 0, 134);
  pack_kernel<<<cdiv(8 * 4 * 32, 256), 256, 0, stream>>>(e_eW2, eE2, 128, 128, 128, 0, 0, 4, 4, 0, 0, 128);
  pack_kernel<<<cdiv(8 * 4 * 32, 256), 256, 0, stream>>>(e_nW1, eN1, 128, 128, 128, 0, 0, 4, 4, 0, 0, 128);
  pack_kernel<<<cdiv(8 * 4 * 32, 256), 256, 0, stream>>>(e_nW2, eN2, 128, 128, 128, 0, 0, 4, 4, 0, 0, 128);
  pack_kernel<<<cdiv(2 * 4 * 32, 256), 256, 0, stream>>>(e_mW, eM, 32, 128, 128, 0, 0, 4, 4, 0, 0, 128);
  pack_kernel<<<cdiv(2 * 4 * 32, 256), 256, 0, stream>>>(e_sW, eS, 32, 128, 128, 0, 0, 4, 4, 0, 0, 128);
  pack_kernel<<<cdiv(8 * 4 * 32, 256), 256, 0, stream>>>(p_eW1, pE1, 128, 128, 256, 0, 0, 4, 4, 1, 0, 128);
  pack_kernel<<<cdiv(8 * 4 * 32, 256), 256, 0, stream>>>(p_eW2, pE2, 128, 128, 128, 0, 0, 4, 4, 0, 0, 128);
  pack_kernel<<<cdiv(8 * 4 * 32, 256), 256, 0, stream>>>(p_nW1, pN1, 128, 128, 128, 0, 0, 4, 4, 0, 0, 128);
  pack_kernel<<<cdiv(8 * 4 * 32, 256), 256, 0, stream>>>(p_nW2, pN2, 128, 128, 128, 0, 0, 4, 4, 0, 0, 128);
  pack_kernel<<<cdiv(2 * 4 * 32, 256), 256, 0, stream>>>(p_mW, pM, 32, 128, 128, 0, 0, 4, 4, 0, 0, 128);
  pack_kernel<<<cdiv(2 * 4 * 32, 256), 256, 0, stream>>>(p_sW, pS, 32, 128, 128, 0, 0, 4, 4, 0, 0, 128);
  pack_kernel<<<cdiv(8 * 5 * 32, 256), 256, 0, stream>>>(dc_eW1, dE1, 128, 160, 320, 0, 0, 5, 5, 1, 0, 160);
  pack_kernel<<<cdiv(8 * 4 * 32, 256), 256, 0, stream>>>(dc_eW2, dE2, 128, 128, 128, 0, 0, 4, 4, 0, 0, 128);
  pack_kernel<<<cdiv(8 * 4 * 32, 256), 256, 0, stream>>>(dc_nW1, dN1, 128, 128, 128, 0, 0, 4, 4, 0, 0, 128);
  pack_kernel<<<cdiv(8 * 4 * 32, 256), 256, 0, stream>>>(dc_nW2, dN2, 128, 128, 128, 0, 0, 4, 4, 0, 0, 128);
  pack_kernel<<<cdiv(2 * 4 * 32, 256), 256, 0, stream>>>(dc_mW, dM, 32, 128, 128, 0, 0, 4, 4, 0, 0, 128);
  pack_kernel<<<cdiv(2 * 4 * 32, 256), 256, 0, stream>>>(dc_sW, dS, 32, 128, 128, 0, 0, 4, 4, 0, 0, 128);
  pack_kernel<<<1, 256, 0, stream>>>(phiW, phiP, 32, 32, 32, 0, 0, 1, 1, 0, 0, 32);

  const int RT = RROWS / 16;  // 176
  for (int t = 0; t < SEQL; ++t) {
    lstm_kernel<<<RT, 256, 0, stream>>>(prev, c1, c2, h1b, h2b, W0p, W1p, bih0, bhh0, bih1, bhh1);
    gn_e_kernel<<<RT, 256, 0, stream>>>(t, 1, target, h2b, eE1, e_eb1, eE2, e_eb2, 5, 160, eEnc);
    gn_p2_kernel<<<8, 256, 0, stream>>>(eEnc, eN1, e_nb1, eN2, e_nb2, eM, e_mb, eS, e_sb, encM, encS);
    if (t == SEQL - 1) {  // prior only feeds the final-step KLD
      gn_e_kernel<<<RT, 256, 0, stream>>>(t, 0, target, h2b, pE1, p_eb1, pE2, p_eb2, 4, 128, ePri);
      gn_p2_kernel<<<8, 256, 0, stream>>>(ePri, pN1, p_nb1, pN2, p_nb2, pM, p_mb, pS, p_sb, priM, priS);
    }
    gn_e_dec_kernel<<<RT, 256, 0, stream>>>(t, eps_enc, encM, encS, h2b, phiP, phib,
                                            dE1, dc_eb1, dE2, dc_eb2, eDec);
    gn_p2_kernel<<<8, 256, 0, stream>>>(eDec, dN1, dc_nb1, dN2, dc_nb2, dM, dc_mb, dS, dc_sb, dmB, dsB);
    out_step_kernel<<<OUTBLKS, 256, 0, stream>>>(t, eps_dec, dmB, dsB, outW, outb,
                                                 input, mask, prev, pred, partials);
  }
  final_kernel<<<1, 256, 0, stream>>>(partials, encM, encS, priM, priS, lossp);
}
